// MLPgenerator_7670811591236
// MI455X (gfx1250) — matrix operand order repaired and checked by reading, not compiled, not run on hardware
//
#include <hip/hip_runtime.h>
#include <hip/hip_bf16.h>
#include <math.h>

// ---------------------------------------------------------------------------
// MLP generator for MI455X (gfx1250, wave32, WMMA).
// Pipeline:
//   k0: zero onehot output (512 MB, b128 stores)  -- memory floor ~22us
//   k1: h0 = lrelu(z @ W0^T + b0)                  [256 x 512]
//   k2: h1 = lrelu(BN(h0 @ W1^T))                  [256 x 1024]  (b1 cancels in BN)
//   k3: split h1 -> bf16 hi/lo (3xBF16 trick, ~f32 accuracy)
//   k4: per-WG 256x32 stripe of gen = h1 @ W2^T via v_wmma_f32_16x16x32_bf16
//       (3 WMMAs per K=32 chunk: hi*hi + hi*lo + lo*hi). W2 staging is
//       double-buffered in LDS with register prefetch of the next chunk
//       (1 barrier per chunk). Local BN stats over the full batch (M=256
//       owned by the WG), per-row local top-5 -> ws.
//   k5: merge 3125x5 candidates per row -> global top-5, scatter 1.0s + idx
// Workspace needed: ~35 MB.
// ---------------------------------------------------------------------------

typedef __attribute__((ext_vector_type(16))) __bf16 v16bf;
typedef __attribute__((ext_vector_type(8)))  __bf16 v8bf;
typedef __attribute__((ext_vector_type(4)))  __bf16 v4bf;
typedef __attribute__((ext_vector_type(8)))  float  v8f;

#define BS      256
#define D0      128
#define D1      512
#define D2      1024
#define NOUT    100000
#define TOPK    5
#define EPSBN   0.8f
#define SLOPE   0.2f
#define NTILE   32                       // gen columns per workgroup
#define NGRP    (NOUT / NTILE)           // 3125 workgroups
#define NCAND   (NGRP * TOPK)            // 15625 candidates per row
#define ONEHOT_ELEMS (256ull * 5ull * 100000ull)

__device__ __forceinline__ bool tk_better(float av, int ai, float bv, int bi) {
    // jax.lax.top_k ordering: larger value wins; ties -> lower index wins.
    return (av > bv) || (av == bv && ai < bi);
}

__device__ __forceinline__ void tk_insert(float* tv, int* ti, float v, int id) {
    if (!tk_better(v, id, tv[4], ti[4])) return;
    tv[4] = v; ti[4] = id;
#pragma unroll
    for (int p = 4; p > 0; --p) {
        if (tk_better(tv[p], ti[p], tv[p - 1], ti[p - 1])) {
            float fv = tv[p]; tv[p] = tv[p - 1]; tv[p - 1] = fv;
            int   fi = ti[p]; ti[p] = ti[p - 1]; ti[p - 1] = fi;
        }
    }
}

// ---------------- k0: zero the one-hot region of d_out ---------------------
__global__ __launch_bounds__(256) void k_zero(float4* __restrict__ out) {
    size_t i = (size_t)blockIdx.x * 256 + threadIdx.x;   // 32e6 float4 total
    out[i] = make_float4(0.f, 0.f, 0.f, 0.f);
}

// ---------------- k1: layer0  h0 = lrelu(z @ W0^T + b0) --------------------
__global__ __launch_bounds__(256) void k_layer0(const float* __restrict__ z,
                                                const float* __restrict__ W0,
                                                const float* __restrict__ b0,
                                                float* __restrict__ h0) {
    __shared__ float zs[D0];
    const int r = blockIdx.x, t = threadIdx.x;
    if (t < D0) zs[t] = z[r * D0 + t];
    __syncthreads();
    for (int c = t; c < D1; c += 256) {
        const float* w = W0 + (size_t)c * D0;
        float a = b0[c];
#pragma unroll 8
        for (int k = 0; k < D0; ++k) a = fmaf(zs[k], w[k], a);
        h0[r * D1 + c] = (a >= 0.f) ? a : SLOPE * a;
    }
}

// ---------------- k2: layer1 fused GEMM + BN + lrelu -----------------------
// 32 blocks x 256 threads; block owns 32 columns x all 256 rows (full batch
// -> BN stats computed locally, b1 skipped since it cancels in BN).
__global__ __launch_bounds__(256) void k_layer1(const float* __restrict__ h0,
                                                const float* __restrict__ W1,
                                                const float* __restrict__ g1,
                                                const float* __restrict__ be1,
                                                float* __restrict__ h1) {
    __shared__ float h0s[BS][32];      // 32 KB
    __shared__ float w1s[32][33];      // padded vs bank conflicts
    __shared__ float ps[8][32], pss[8][32];
    __shared__ float scl[32], sft[32];

    const int t = threadIdx.x;
    const int c = t & 31, seg = t >> 5;       // col, row-segment (wave==seg)
    const int cg = blockIdx.x * 32;

    float acc[32];
#pragma unroll
    for (int i = 0; i < 32; ++i) acc[i] = 0.f;

    for (int k0 = 0; k0 < D1; k0 += 32) {
        // stage h0[256][k0..k0+32] : 2048 float4, 8 per thread
#pragma unroll
        for (int q = 0; q < 8; ++q) {
            int e = t + q * 256;                 // 0..2047
            int row = e >> 3, kk = (e & 7) * 4;
            float4 v = *(const float4*)(h0 + (size_t)row * D1 + k0 + kk);
            h0s[row][kk] = v.x; h0s[row][kk + 1] = v.y;
            h0s[row][kk + 2] = v.z; h0s[row][kk + 3] = v.w;
        }
        { // stage W1[cg..cg+32][k0..k0+32] transposed: 256 float4, 1/thread
            int cc = t >> 3, kk = (t & 7) * 4;
            float4 v = *(const float4*)(W1 + (size_t)(cg + cc) * D1 + k0 + kk);
            w1s[kk][cc] = v.x; w1s[kk + 1][cc] = v.y;
            w1s[kk + 2][cc] = v.z; w1s[kk + 3][cc] = v.w;
        }
        __syncthreads();
#pragma unroll 4
        for (int kk = 0; kk < 32; ++kk) {
            float w = w1s[kk][c];
#pragma unroll
            for (int i = 0; i < 32; ++i)
                acc[i] = fmaf(h0s[seg * 32 + i][kk], w, acc[i]);
        }
        __syncthreads();
    }
    // BN stats over the 256-row batch (b1 cancels exactly)
    float s = 0.f, ss = 0.f;
#pragma unroll
    for (int i = 0; i < 32; ++i) { s += acc[i]; ss += acc[i] * acc[i]; }
    ps[seg][c] = s; pss[seg][c] = ss;
    __syncthreads();
    if (t < 32) {
        float S = 0.f, SS = 0.f;
#pragma unroll
        for (int g = 0; g < 8; ++g) { S += ps[g][t]; SS += pss[g][t]; }
        float m  = S * (1.f / 256.f);
        float vv = SS * (1.f / 256.f) - m * m;
        float sc = g1[cg + t] * rsqrtf(vv + EPSBN);
        scl[t] = sc; sft[t] = be1[cg + t] - m * sc;
    }
    __syncthreads();
    const float sc = scl[c], sh = sft[c];
#pragma unroll
    for (int i = 0; i < 32; ++i) {
        float v = acc[i] * sc + sh;
        v = (v >= 0.f) ? v : SLOPE * v;
        h1[(size_t)(seg * 32 + i) * D2 + cg + c] = v;
    }
}

// ---------------- k3: split h1 into bf16 hi/lo -----------------------------
__global__ __launch_bounds__(256) void k_split(const float* __restrict__ h1,
                                               __bf16* __restrict__ Ahi,
                                               __bf16* __restrict__ Alo) {
    int e = (blockIdx.x * 256 + threadIdx.x) * 4;     // 262144 total
    float4 v = *(const float4*)(h1 + e);
    float f[4] = {v.x, v.y, v.z, v.w};
    v4bf hi, lo;
#pragma unroll
    for (int q = 0; q < 4; ++q) {
        __bf16 h = (__bf16)f[q];
        hi[q] = h;
        lo[q] = (__bf16)(f[q] - (float)h);
    }
    *(v4bf*)(Ahi + e) = hi;
    *(v4bf*)(Alo + e) = lo;
}

// ---------------- k4 helpers ----------------------------------------------
__device__ __forceinline__ void stage_w2(const float4 w, int c, int kk,
                                         __bf16 (*bh)[32], __bf16 (*bl)[32]) {
    float f[4] = {w.x, w.y, w.z, w.w};
#pragma unroll
    for (int q = 0; q < 4; ++q) {
        __bf16 h = (__bf16)f[q];
        bh[c][kk + q] = h;
        bl[c][kk + q] = (__bf16)(f[q] - (float)h);
    }
}

__device__ __forceinline__ void compute_chunk(int kc,
        const __bf16 (*Bh)[32], const __bf16 (*Bl)[32],
        const __bf16* __restrict__ Ahi, const __bf16* __restrict__ Alo,
        v8f* acc, int lane, int mb, int ng) {
    // B fragments (B 32x16, k order used here: lane<16 -> K0..15, lane>=16 -> K16..31,
    // contiguous per lane; col = lane%16)
    const int bc = ng * 16 + (lane & 15);
    const int bk = (lane < 16) ? 0 : 16;
    v16bf bh = *(const v16bf*)&Bh[bc][bk];
    v16bf bl = *(const v16bf*)&Bl[bc][bk];
    // A fragments per m-tile (A 16x32: lane<16 K0-15;
    // lane>=16 K16-31, contiguous to match B), 3xBF16 accumulation
#pragma unroll
    for (int i = 0; i < 4; ++i) {
        const int row = (mb + 4 * i) * 16 + (lane & 15);
        const int ka  = kc + ((lane < 16) ? 0 : 16);
        const __bf16* ph = Ahi + (size_t)row * D2 + ka;
        const __bf16* pl = Alo + (size_t)row * D2 + ka;
        v8bf ah0 = *(const v8bf*)ph, ah1 = *(const v8bf*)(ph + 8);
        v8bf al0 = *(const v8bf*)pl, al1 = *(const v8bf*)(pl + 8);
        v16bf ah = __builtin_shufflevector(ah0, ah1,
            0,1,2,3,4,5,6,7,8,9,10,11,12,13,14,15);
        v16bf al = __builtin_shufflevector(al0, al1,
            0,1,2,3,4,5,6,7,8,9,10,11,12,13,14,15);
        acc[i] = __builtin_amdgcn_wmma_f32_16x16x32_bf16(
                     false, ah, false, bh, (short)0, acc[i], false, false);
        acc[i] = __builtin_amdgcn_wmma_f32_16x16x32_bf16(
                     false, ah, false, bl, (short)0, acc[i], false, false);
        acc[i] = __builtin_amdgcn_wmma_f32_16x16x32_bf16(
                     false, al, false, bh, (short)0, acc[i], false, false);
    }
}

// ---------------- k4: GEMM2 stripe + local BN + local top-5 ----------------
// 3125 blocks x 256 threads (8 waves). Block owns gen[:, j0..j0+32].
// Wave w: n-group = w&1 (16 cols), m-tiles = (w>>1) + 4i, i=0..3.
// Double-buffered B staging: global W2 chunk prefetched into registers a
// full compute-phase ahead; one barrier per K=32 chunk.
__global__ __launch_bounds__(256) void k_gemm2(const float* __restrict__ W2,
                                               const __bf16* __restrict__ Ahi,
                                               const __bf16* __restrict__ Alo,
                                               const float* __restrict__ g2,
                                               const float* __restrict__ be2,
                                               float* __restrict__ cand_v,
                                               int* __restrict__ cand_i) {
    __shared__ __bf16 Bhi[2][NTILE][32];       // double-buffered, 4 KB
    __shared__ __bf16 Blo[2][NTILE][32];       // 4 KB
    __shared__ float  tile[BS][NTILE + 1];     // padded, ~34 KB
    __shared__ float  ps[8][NTILE], pss[8][NTILE];
    __shared__ float  scl[NTILE], sft[NTILE];

    const int t    = threadIdx.x;
    const int wave = t >> 5, lane = t & 31;
    const int j0   = blockIdx.x * NTILE;
    const int ng   = wave & 1;
    const int mb   = wave >> 1;

    v8f acc[4] = {};   // f32 accumulators, 4 m-tiles x 16x16

    const int sc_c  = t >> 3;                  // staging: col 0..31
    const int sc_kk = (t & 7) * 4;             // k 0,4,..28
    const float* wsrc = W2 + (size_t)(j0 + sc_c) * D2 + sc_kk;

    float4 wA = *(const float4*)wsrc;          // preload chunk 0
    for (int kc = 0; kc < D2; kc += 64) {
        // ---- stage chunk kc into buffer 0; prefetch chunk kc+32 ----
        stage_w2(wA, sc_c, sc_kk, Bhi[0], Blo[0]);
        float4 wB = *(const float4*)(wsrc + kc + 32);
        if (kc + 64 < D2) __builtin_prefetch(wsrc + kc + 64, 0, 1);
        __syncthreads();
        compute_chunk(kc, Bhi[0], Blo[0], Ahi, Alo, acc, lane, mb, ng);
        // ---- stage chunk kc+32 into buffer 1; prefetch chunk kc+64 ----
        stage_w2(wB, sc_c, sc_kk, Bhi[1], Blo[1]);
        if (kc + 64 < D2) wA = *(const float4*)(wsrc + kc + 64);
        __syncthreads();
        compute_chunk(kc + 32, Bhi[1], Blo[1], Ahi, Alo, acc, lane, mb, ng);
        // safety: stage(i+2) into buf0 happens after the barrier that follows
        // stage(i+1), which in turn follows every thread's compute(i) -> safe.
    }

    // ---- spill accumulators to LDS tile (C layout: lane<16 M=v, lane>=16 M=8+v;
    //      N = lane%16). b2 skipped: cancels in BN. ----
    __syncthreads();
#pragma unroll
    for (int i = 0; i < 4; ++i) {
        const int row0 = (mb + 4 * i) * 16 + ((lane < 16) ? 0 : 8);
        const int col  = ng * 16 + (lane & 15);
#pragma unroll
        for (int v = 0; v < 8; ++v) tile[row0 + v][col] = acc[i][v];
    }
    __syncthreads();
    // ---- BN statistics over full batch (256 rows owned by this WG) ----
    {
        const int c = t & 31, seg = t >> 5;
        float s = 0.f, ss = 0.f;
        for (int r = seg * 32; r < seg * 32 + 32; ++r) {
            float x = tile[r][c]; s += x; ss += x * x;
        }
        ps[seg][c] = s; pss[seg][c] = ss;
    }
    __syncthreads();
    if (t < 32) {
        float S = 0.f, SS = 0.f;
#pragma unroll
        for (int g = 0; g < 8; ++g) { S += ps[g][t]; SS += pss[g][t]; }
        float m  = S * (1.f / 256.f);
        float vv = SS * (1.f / 256.f) - m * m;
        float sc = g2[j0 + t] * rsqrtf(vv + EPSBN);
        scl[t] = sc; sft[t] = be2[j0 + t] - m * sc;
    }
    __syncthreads();
    // ---- per-row local top-5 over this WG's 32 columns ----
    {
        const int r = t;
        float tv[5]; int ti[5];
#pragma unroll
        for (int k = 0; k < 5; ++k) { tv[k] = -__builtin_inff(); ti[k] = 0x7fffffff; }
        for (int c = 0; c < NTILE; ++c) {
            float v = tile[r][c] * scl[c] + sft[c];
            tk_insert(tv, ti, v, j0 + c);
        }
        size_t base = (size_t)r * NCAND + (size_t)blockIdx.x * TOPK;
#pragma unroll
        for (int k = 0; k < 5; ++k) { cand_v[base + k] = tv[k]; cand_i[base + k] = ti[k]; }
    }
}

// ---------------- k5: merge candidates -> global top-5, scatter output -----
__global__ __launch_bounds__(256) void k_merge(const float* __restrict__ cand_v,
                                               const int* __restrict__ cand_i,
                                               float* __restrict__ out) {
    __shared__ float mv[256][5];
    __shared__ int   mi[256][5];
    const int r = blockIdx.x, t = threadIdx.x;
    const float* cv = cand_v + (size_t)r * NCAND;
    const int*   ci = cand_i + (size_t)r * NCAND;

    float tv[5]; int ti[5];
#pragma unroll
    for (int k = 0; k < 5; ++k) { tv[k] = -__builtin_inff(); ti[k] = 0x7fffffff; }
    for (int e = t; e < NCAND; e += 256) tk_insert(tv, ti, cv[e], ci[e]);
#pragma unroll
    for (int k = 0; k < 5; ++k) { mv[t][k] = tv[k]; mi[t][k] = ti[k]; }
    __syncthreads();
    for (int off = 128; off > 0; off >>= 1) {
        if (t < off) {
#pragma unroll
            for (int k = 0; k < 5; ++k)
                tk_insert(tv, ti, mv[t + off][k], mi[t + off][k]);
#pragma unroll
            for (int k = 0; k < 5; ++k) { mv[t][k] = tv[k]; mi[t][k] = ti[k]; }
        }
        __syncthreads();
    }
    if (t == 0) {
#pragma unroll
        for (int k = 0; k < 5; ++k) {
            int idx = ti[k];
            out[(size_t)(r * TOPK + k) * NOUT + idx] = 1.0f;   // straight-through fwd = 1
            out[ONEHOT_ELEMS + (size_t)r * TOPK + k] = (float)idx;
        }
    }
}

// ---------------------------------------------------------------------------
extern "C" void kernel_launch(void* const* d_in, const int* in_sizes, int n_in,
                              void* d_out, int out_size, void* d_ws, size_t ws_size,
                              hipStream_t stream) {
    const float* z   = (const float*)d_in[0];
    const float* W0  = (const float*)d_in[1];
    const float* b0  = (const float*)d_in[2];
    const float* W1  = (const float*)d_in[3];
    // d_in[4] = b1 : cancels in BN, unused
    const float* g1  = (const float*)d_in[5];
    const float* be1 = (const float*)d_in[6];
    const float* W2  = (const float*)d_in[7];
    // d_in[8] = b2 : cancels in BN, unused
    const float* g2  = (const float*)d_in[9];
    const float* be2 = (const float*)d_in[10];

    // workspace carve-up (~35 MB)
    float*  h0     = (float*)d_ws;                       // 131072 f
    float*  h1     = h0 + BS * D1;                       // 262144 f
    __bf16* Ahi    = (__bf16*)(h1 + BS * D2);            // 262144 bf16
    __bf16* Alo    = Ahi + BS * D2;                      // 262144 bf16
    float*  cand_v = (float*)(Alo + BS * D2);            // 4,000,000 f
    int*    cand_i = (int*)(cand_v + (size_t)BS * NCAND);// 4,000,000 i

    float* out = (float*)d_out;

    k_zero  <<<(int)(ONEHOT_ELEMS / 4 / 256), 256, 0, stream>>>((float4*)out);
    k_layer0<<<BS,   256, 0, stream>>>(z, W0, b0, h0);
    k_layer1<<<D2/32,256, 0, stream>>>(h0, W1, g1, be1, h1);
    k_split <<<BS,   256, 0, stream>>>(h1, Ahi, Alo);
    k_gemm2 <<<NGRP, 256, 0, stream>>>(W2, Ahi, Alo, g2, be2, cand_v, cand_i);
    k_merge <<<BS,   256, 0, stream>>>(cand_v, cand_i, out);
}
